// StochasticKeyNet_54778012893669
// MI455X (gfx1250) — matrix operand order repaired and checked by reading, not compiled, not run on hardware
//
#include <hip/hip_runtime.h>
#include <hip/hip_bf16.h>

typedef __attribute__((ext_vector_type(16))) __bf16 v16bf;
typedef __attribute__((ext_vector_type(8)))  float  v8f;

__device__ __forceinline__ unsigned short f2bf(float f) {
    unsigned int u = __float_as_uint(f);
    return (unsigned short)((u + 0x8000u) >> 16);   // round-to-nearest (ties away)
}
__device__ __forceinline__ float bf2f(unsigned short s) {
    return __uint_as_float(((unsigned int)s) << 16);
}
__device__ __forceinline__ unsigned pk2(float a, float b) {
    return (unsigned)f2bf(a) | ((unsigned)f2bf(b) << 16);
}

// ---------------------------------------------------------------------------
// Layout conversions.
// Activations: NHWC bf16.  x: [256][3][32][32] f32 -> [256][32*32][3] bf16
// ---------------------------------------------------------------------------
__global__ __launch_bounds__(256) void cvt_x(const float* __restrict__ src,
                                             unsigned short* __restrict__ dst) {
    int i = blockIdx.x * 256 + threadIdx.x;            // 786432 total
    int b = i / 3072, r = i - b * 3072;
    int c = r >> 10, p = r & 1023;
    dst[(b * 1024 + p) * 3 + c] = f2bf(src[i]);
}
// Weights: [O][CIN][3][3] f32 -> [O][9*CIN] bf16 with k = (ky*3+kx)*CIN + ci
__global__ __launch_bounds__(256) void cvt_w(const float* __restrict__ src,
                                             unsigned short* __restrict__ dst,
                                             int O, int CIN) {
    int i = blockIdx.x * 256 + threadIdx.x;
    int n = O * CIN * 9;
    if (i < n) {
        int o  = i / (CIN * 9);
        int r  = i - o * CIN * 9;
        int ci = r / 9;
        int g  = r - ci * 9;
        dst[(o * 9 + g) * CIN + ci] = f2bf(src[i]);
    }
}

// ---------------------------------------------------------------------------
// Implicit-GEMM 3x3 conv (pad 1), NHWC bf16, WMMA f32_16x16x32_bf16.
//   M = COUT, N = B*HO*WO, K = 9*CIN  (k = (ky*3+kx)*CIN + ci)
//   256 threads = 8 waves (2M x 4N); wave tile 32x32; block tile 64 x 128.
//   K-slab of 64 per barrier pair -> 8 WMMA / wave / iteration.
//   Staging via GLOBAL_LOAD_ASYNC_TO_LDS_B128 when CIN % 32 == 0.
// ---------------------------------------------------------------------------
template<int CIN, int H, int W, int COUT, int STRIDE>
__global__ __launch_bounds__(256)
void conv3x3_wmma(const unsigned short* __restrict__ act,   // [B][H][W][CIN]  bf16
                  const unsigned short* __restrict__ wgt,   // [COUT][9*CIN]   bf16
                  const float* __restrict__ bias,           // [COUT] f32
                  unsigned short* __restrict__ out) {       // [B][HO][WO][COUT] bf16
    constexpr int HOUT   = H / STRIDE;
    constexpr int WOUT   = W / STRIDE;
    constexpr int K      = 9 * CIN;
    constexpr int KB     = 64;
    constexpr int KSTEPS = (K + KB - 1) / KB;
    constexpr bool NICE  = (CIN % 32 == 0);      // => K % 32 == 0, aligned chunks

    __shared__ __align__(16) unsigned short As[64 * KB];    // [m][k]   8 KB
    __shared__ __align__(16) unsigned short Bs[128 * KB];   // [n][k]  16 KB

    const int t    = threadIdx.x;
    const int lane = t & 31;
    const int wave = t >> 5;
    const int wm   = wave >> 2;          // 0..1  (M)
    const int wn   = wave & 3;           // 0..3  (N)
    const int hx   = lane >> 4;          // lane half (ISA fragment layout)
    const int l16  = lane & 15;

    const int n0 = blockIdx.x * 128;
    const int m0 = blockIdx.y * 64;

    // A staging: thread -> (m, 16 k's) = 32B = 2 async b128
    const int am  = t >> 2;
    const int akq = (t & 3) * 16;
    // B staging: thread -> (n, 32 k's) = 64B = 4 async b128
    const int bn  = t >> 1;
    const int bkq = (t & 1) * 32;
    const int ncol = n0 + bn;
    const int img  = ncol / (HOUT * WOUT);
    const int rem  = ncol % (HOUT * WOUT);
    const int oy   = rem / WOUT;
    const int ox   = rem % WOUT;

    const unsigned a_lds = (unsigned)(size_t)&As[am * KB + akq];
    const unsigned b_lds = (unsigned)(size_t)&Bs[bn * KB + bkq];
    const int      mg    = m0 + am;

    v8f acc[2][2];
    #pragma unroll
    for (int i = 0; i < 2; ++i)
        #pragma unroll
        for (int j = 0; j < 2; ++j)
            acc[i][j] = v8f{0.f,0.f,0.f,0.f,0.f,0.f,0.f,0.f};

    for (int ks = 0; ks < KSTEPS; ++ks) {
        const int kbase = ks * KB;

        if constexpr (NICE) {
            // ---- weights: 2 async b128 (chunk of 16 k, wholly in or out) ----
            if (mg < COUT && (K % KB == 0 || kbase + akq < K)) {
                const unsigned short* gp = wgt + mg * K + kbase + akq;
                asm volatile("global_load_async_to_lds_b128 %0, %1, off"
                             :: "v"(a_lds), "v"(gp) : "memory");
                asm volatile("global_load_async_to_lds_b128 %0, %1, off offset:16"
                             :: "v"(a_lds), "v"(gp) : "memory");
            } else {
                *(uint4*)&As[am * KB + akq]     = uint4{0, 0, 0, 0};
                *(uint4*)&As[am * KB + akq + 8] = uint4{0, 0, 0, 0};
            }
            // ---- im2col: 32 consecutive k = 32 contiguous channels (one group) ----
            const int g  = (kbase + bkq) / CIN;       // ky*3+kx
            const int ci = (kbase + bkq) % CIN;
            const int ky = g / 3, kx = g - ky * 3;
            const int iy = oy * STRIDE - 1 + ky;
            const int ix = ox * STRIDE - 1 + kx;
            if (iy >= 0 && iy < H && ix >= 0 && ix < W &&
                (K % KB == 0 || kbase + bkq < K)) {
                const unsigned short* gp = act + ((img * H + iy) * W + ix) * CIN + ci;
                asm volatile("global_load_async_to_lds_b128 %0, %1, off"
                             :: "v"(b_lds), "v"(gp) : "memory");
                asm volatile("global_load_async_to_lds_b128 %0, %1, off offset:16"
                             :: "v"(b_lds), "v"(gp) : "memory");
                asm volatile("global_load_async_to_lds_b128 %0, %1, off offset:32"
                             :: "v"(b_lds), "v"(gp) : "memory");
                asm volatile("global_load_async_to_lds_b128 %0, %1, off offset:48"
                             :: "v"(b_lds), "v"(gp) : "memory");
            } else {
                #pragma unroll
                for (int q = 0; q < 4; ++q)
                    *(uint4*)&Bs[bn * KB + bkq + 8 * q] = uint4{0, 0, 0, 0};
            }
            asm volatile("s_wait_asynccnt 0x0" ::: "memory");
        } else {
            // ---- scalar fallback (layer 1, CIN=3, K=27, KSTEPS=1) ----
            #pragma unroll
            for (int j = 0; j < 16; ++j) {
                const int kg = kbase + akq + j;
                unsigned short v = 0;
                if (mg < COUT && kg < K) v = wgt[mg * K + kg];
                As[am * KB + akq + j] = v;
            }
            #pragma unroll
            for (int j = 0; j < 32; ++j) {
                const int kg = kbase + bkq + j;
                const int g  = kg / CIN;
                const int ci = kg - g * CIN;
                const int ky = g / 3, kx = g - ky * 3;
                const int iy = oy * STRIDE - 1 + ky;
                const int ix = ox * STRIDE - 1 + kx;
                unsigned short v = 0;
                if (kg < K && iy >= 0 && iy < H && ix >= 0 && ix < W)
                    v = act[((img * H + iy) * W + ix) * CIN + ci];
                Bs[bn * KB + bkq + j] = v;
            }
        }
        __syncthreads();

        // ---- fragments in ISA lane layout (all ds_load_b128), 2 K-subtiles ----
        union { uint4 u[2]; v16bf v; } fa[2][2], fb[2][2];
        #pragma unroll
        for (int i = 0; i < 2; ++i) {
            const int mrow = wm * 32 + i * 16 + l16;
            #pragma unroll
            for (int s = 0; s < 2; ++s) {
                fa[i][s].u[0] = *(const uint4*)&As[mrow * KB + s * 32 + hx * 8];
                fa[i][s].u[1] = *(const uint4*)&As[mrow * KB + s * 32 + 16 + hx * 8];
            }
        }
        #pragma unroll
        for (int j = 0; j < 2; ++j) {
            const int nrow = wn * 32 + j * 16 + l16;
            #pragma unroll
            for (int s = 0; s < 2; ++s) {
                fb[j][s].u[0] = *(const uint4*)&Bs[nrow * KB + s * 32 + hx * 8];
                fb[j][s].u[1] = *(const uint4*)&Bs[nrow * KB + s * 32 + 16 + hx * 8];
            }
        }
        #pragma unroll
        for (int s = 0; s < 2; ++s)
            #pragma unroll
            for (int i = 0; i < 2; ++i)
                #pragma unroll
                for (int j = 0; j < 2; ++j)
                    acc[i][j] = __builtin_amdgcn_wmma_f32_16x16x32_bf16(
                                    false, fa[i][s].v, false, fb[j][s].v,
                                    (short)0, acc[i][j], false, false);
        __syncthreads();
    }

    // ---- epilogue: bias + ReLU; 8 acc rows = 8 consecutive channels (NHWC) ----
    #pragma unroll
    for (int j = 0; j < 2; ++j) {
        const int pix = n0 + wn * 32 + j * 16 + l16;          // [B*HO*WO] pixel
        #pragma unroll
        for (int i = 0; i < 2; ++i) {
            const int mbase = m0 + wm * 32 + i * 16 + hx * 8; // D: VGPR r -> M=r+8*hx
            float v[8];
            #pragma unroll
            for (int r = 0; r < 8; ++r) {
                float z = acc[i][j][r] + ((COUT % 8 == 0) ? bias[mbase + r]
                                          : (mbase + r < COUT ? bias[mbase + r] : 0.f));
                v[r] = z > 0.f ? z : 0.f;
            }
            if constexpr (COUT % 8 == 0) {
                if (mbase < COUT) {
                    uint4 o = { pk2(v[0], v[1]), pk2(v[2], v[3]),
                                pk2(v[4], v[5]), pk2(v[6], v[7]) };
                    *(uint4*)&out[(size_t)pix * COUT + mbase] = o;
                }
            } else {
                #pragma unroll
                for (int r = 0; r < 8; ++r)
                    if (mbase + r < COUT)
                        out[(size_t)pix * COUT + mbase + r] = f2bf(v[r]);
            }
        }
    }
}

// ---------------------------------------------------------------------------
// 8x8 mean pool of relu(z9) (NHWC [256][64][10]) + final key:
// out[b][i] = d10[i] * aug(p)[perm10[i]],  aug(p) = [p0..p9, 1]
// ---------------------------------------------------------------------------
__global__ __launch_bounds__(32)
void pool_key(const unsigned short* __restrict__ a9,
              const int* __restrict__ perm, const float* __restrict__ dv,
              float* __restrict__ out) {
    __shared__ float p[10];
    const int b = blockIdx.x;
    const int t = threadIdx.x;
    if (t < 10) {
        const unsigned short* base = a9 + b * 640 + t;
        float s = 0.f;
        #pragma unroll
        for (int i = 0; i < 64; ++i) s += bf2f(base[i * 10]);
        p[t] = s * (1.f / 64.f);
    }
    __syncthreads();
    if (t < 11) {
        const int pi = perm[t];
        out[b * 11 + t] = dv[t] * ((pi < 10) ? p[pi] : 1.0f);
    }
}

// ---------------------------------------------------------------------------
extern "C" void kernel_launch(void* const* d_in, const int* in_sizes, int n_in,
                              void* d_out, int out_size, void* d_ws, size_t ws_size,
                              hipStream_t stream) {
    (void)in_sizes; (void)n_in; (void)out_size; (void)ws_size;
    constexpr int B = 256;
    constexpr size_t ACT_ELEMS = (size_t)B * 96 * 32 * 32;   // max layer, 25,165,824

    unsigned short* actA = (unsigned short*)d_ws;
    unsigned short* actB = actA + ACT_ELEMS;
    unsigned short* wbuf = actB + ACT_ELEMS;

    const float* wsrc[9]; const float* bsrc[9];
    for (int l = 0; l < 9; ++l) {
        wsrc[l] = (const float*)d_in[1 + 2 * l];
        bsrc[l] = (const float*)d_in[2 + 2 * l];
    }
    const int Ocnt[9] = { 96, 96, 96, 192, 192, 192, 192, 192, 10 };
    const int Icnt[9] = {  3, 96, 96,  96, 192, 192, 192, 192, 192 };
    unsigned short* wbf[9];
    { unsigned short* p = wbuf;
      for (int l = 0; l < 9; ++l) { wbf[l] = p; p += Ocnt[l] * Icnt[l] * 9; } }

    // convert input (NCHW->NHWC) + weights (OIHW -> [O][9*CIN] reordered) to bf16
    cvt_x<<<B * 3 * 32 * 32 / 256, 256, 0, stream>>>((const float*)d_in[0], actA);
    for (int l = 0; l < 9; ++l) {
        int n = Ocnt[l] * Icnt[l] * 9;
        cvt_w<<<(n + 255) / 256, 256, 0, stream>>>(wsrc[l], wbf[l], Ocnt[l], Icnt[l]);
    }

    // conv chain, grid = (N/128, ceil(COUT/64))
    conv3x3_wmma<  3,32,32, 96,1><<<dim3(B*32*32/128, 2), 256, 0, stream>>>(actA, wbf[0], bsrc[0], actB);
    conv3x3_wmma< 96,32,32, 96,1><<<dim3(B*32*32/128, 2), 256, 0, stream>>>(actB, wbf[1], bsrc[1], actA);
    conv3x3_wmma< 96,32,32, 96,2><<<dim3(B*16*16/128, 2), 256, 0, stream>>>(actA, wbf[2], bsrc[2], actB);
    conv3x3_wmma< 96,16,16,192,1><<<dim3(B*16*16/128, 3), 256, 0, stream>>>(actB, wbf[3], bsrc[3], actA);
    conv3x3_wmma<192,16,16,192,1><<<dim3(B*16*16/128, 3), 256, 0, stream>>>(actA, wbf[4], bsrc[4], actB);
    conv3x3_wmma<192,16,16,192,2><<<dim3(B* 8* 8/128, 3), 256, 0, stream>>>(actB, wbf[5], bsrc[5], actA);
    conv3x3_wmma<192, 8, 8,192,1><<<dim3(B* 8* 8/128, 3), 256, 0, stream>>>(actA, wbf[6], bsrc[6], actB);
    conv3x3_wmma<192, 8, 8,192,1><<<dim3(B* 8* 8/128, 3), 256, 0, stream>>>(actB, wbf[7], bsrc[7], actA);
    conv3x3_wmma<192, 8, 8, 10,1><<<dim3(B* 8* 8/128, 1), 256, 0, stream>>>(actA, wbf[8], bsrc[8], actB);

    // pool + final key (perm10 = d_in[37], d10 = d_in[38])
    pool_key<<<B, 32, 0, stream>>>(actB, (const int*)d_in[37], (const float*)d_in[38],
                                   (float*)d_out);
}
